// Aggregate_32272384262663
// MI455X (gfx1250) — matrix operand order repaired and checked by reading, not compiled, not run on hardware
//
#include <hip/hip_runtime.h>

// ============================================================================
// Self-attention block for MI455X (gfx1250), bf16 WMMA + fp32 accumulate,
// TDM (tensor_load_to_lds) staging of Q/K/V tiles into LDS.
//   B=8, C=256, N=64, K=256.
// Workspace: Q [b][n][pos][c], K [b][n][pos][c], V [b][n][c][pos]  (bf16),
//            then 4 weight matrices converted to bf16 [o][c].
//   3 x 33,554,432 x 2B + 4 x 65,536 x 2B = 192 MiB + 512 KiB.
// ============================================================================

#define B_   8
#define C_   256
#define N_   64
#define K_   256
#define LDSTR 260   // padded LDS row stride (elements); 130 dwords % 64 banks = 2

typedef __attribute__((ext_vector_type(16))) __bf16 v16bf;
typedef __attribute__((ext_vector_type(8)))  float  v8f;
typedef __attribute__((ext_vector_type(4)))  unsigned int v4u;
typedef __attribute__((ext_vector_type(8)))  int    v8i;
typedef __attribute__((ext_vector_type(4)))  int    v4i;

struct __align__(16) U4 { unsigned int x0, x1, x2, x3; };

union BF16x16 {
    v16bf v;
    unsigned int u[8];
    unsigned short s[16];
};

__device__ __forceinline__ unsigned short f2bf(float f) {
    union { float f; unsigned int u; } c; c.f = f;
    unsigned int r = c.u + 0x7FFFu + ((c.u >> 16) & 1u);   // round-to-nearest-even
    return (unsigned short)(r >> 16);
}

// Packed fp32x2 -> bf16x2 (one dword). Uses v_cvt_pk_bf16_f32 when available.
#if __has_builtin(__builtin_amdgcn_cvt_pk_bf16_f32)
typedef __attribute__((ext_vector_type(2))) __bf16 v2bf;
__device__ __forceinline__ unsigned int pack2bf(float lo, float hi) {
    union { v2bf v; unsigned int u; } c;
    c.v = __builtin_amdgcn_cvt_pk_bf16_f32(lo, hi);
    return c.u;
}
#else
__device__ __forceinline__ unsigned int pack2bf(float lo, float hi) {
    return (unsigned int)f2bf(lo) | ((unsigned int)f2bf(hi) << 16);
}
#endif

__device__ __forceinline__ v8f vzero() {
    v8f z;
#pragma unroll
    for (int i = 0; i < 8; ++i) z[i] = 0.0f;
    return z;
}

// A-matrix fragment (bf16 16x32, MxK) from row-major [m][k] buffer.
__device__ __forceinline__ v16bf load_a_frag(const unsigned short* base, int stride, int lane) {
    BF16x16 fr;
    int h = lane >> 4, m = lane & 15;
#pragma unroll
    for (int v = 0; v < 8; ++v) {
        int k0 = ((v & 4) << 2) + h * 8 + ((v & 3) << 1); // (v<4?0:16)+h*8+2*(v&3)
        fr.u[v] = *(const unsigned int*)(base + m * stride + k0);
    }
    return fr.v;
}

// B-matrix fragment (bf16 32x16, KxN) from a buffer laid out [n][k] row-major
// (LDS or global bf16; weights are [o][c] row-major so this serves both).
__device__ __forceinline__ v16bf load_b_frag(const unsigned short* base, int stride, int lane) {
    BF16x16 fr;
    int h = lane >> 4, n = lane & 15;
#pragma unroll
    for (int v = 0; v < 8; ++v) {
        int k0 = ((v & 4) << 2) + h * 8 + ((v & 3) << 1);
        fr.u[v] = *(const unsigned int*)(base + n * stride + k0);
    }
    return fr.v;
}

// ============================================================================
// TDM: async 2D tile copy global(bf16, rows of 512B) -> LDS (row stride 520B).
// D# per CDNA5 ISA ch.8: data_size=4B units, tile_dim0=128, pad 2 dwords per
// 128 dwords (pad_interval=6, pad_amount=1) -> LDS stride 260 shorts.
// Issued by one wave; completion via s_wait_tensorcnt; others sync at barrier.
// ============================================================================
#if __has_builtin(__builtin_amdgcn_tensor_load_to_lds) && \
    __has_builtin(__builtin_amdgcn_s_wait_tensorcnt)
#define HAVE_TDM 1
#endif

__device__ __forceinline__ unsigned int lds_off32(const void* p) {
    // generic pointer to LDS: addr[31:0] is the LDS byte offset (ISA 10.2)
    return (unsigned int)(unsigned long long)p;
}

#ifdef HAVE_TDM
__device__ __forceinline__ void tdm_load_rows(const unsigned short* gsrc,
                                              const unsigned short* ldsdst,
                                              int rows) {
    unsigned long long ga = (unsigned long long)(const void*)gsrc;
    v4u g0;
    g0[0] = 1u;                                            // count=1, user mode
    g0[1] = lds_off32(ldsdst);                             // lds_addr (bytes)
    g0[2] = (unsigned int)ga;                              // global_addr[31:0]
    g0[3] = (unsigned int)((ga >> 32) & 0x01FFFFFFu)       // global_addr[56:32]
          | (2u << 30);                                    // type=2 (image)
    v8i g1;
    g1[0] = (2 << 16)      // data_size = 4B units
          | (1 << 20)      // pad_enable
          | (6 << 22)      // pad_interval: 128 dwords
          | (1 << 25);     // pad_amount: 2 dwords
    g1[1] = (128 << 16);                 // tensor_dim0[15:0] = 128 units (512B row)
    g1[2] = ((rows & 0xFFFF) << 16);     // tensor_dim0[31:16]=0 | tensor_dim1[15:0]
    g1[3] = (128 << 16);                 // tensor_dim1[31:16]=0 | tile_dim0 = 128
    g1[4] = (rows & 0xFFFF);             // tile_dim1 = rows | tile_dim2 = 0
    g1[5] = 128;                         // tensor_dim0_stride = 128 units (dense)
    g1[6] = 0;
    g1[7] = 0;
    v4i z4 = {0, 0, 0, 0};
#if defined(__clang_major__) && (__clang_major__ >= 23)
    v8i z8 = {0, 0, 0, 0, 0, 0, 0, 0};
    __builtin_amdgcn_tensor_load_to_lds(g0, g1, z4, z4, z8, 0);
#else
    __builtin_amdgcn_tensor_load_to_lds(g0, g1, z4, z4, 0);
#endif
}
#endif

// Fallback cooperative copy: rows of 128 dwords -> stride-260 LDS.
__device__ __forceinline__ void coop_copy_rows(const unsigned int* gsrc32,
                                               unsigned short* ldsdst,
                                               int rows, int t) {
    const int total = rows * 128;
#pragma unroll 4
    for (int idx = t; idx < total; idx += 256) {
        const int row  = idx >> 7;
        const int col2 = idx & 127;
        *(unsigned int*)&ldsdst[row * LDSTR + col2 * 2] = gsrc32[idx];
    }
}

__device__ __forceinline__ void tile_load(const unsigned short* gsrc,
                                          unsigned short* ldsdst,
                                          int rows, int t, int w) {
#ifdef HAVE_TDM
    if (w == 0) {
        tdm_load_rows(gsrc, ldsdst, rows);
        __builtin_amdgcn_s_wait_tensorcnt(0);
    }
#else
    (void)w;
    coop_copy_rows((const unsigned int*)gsrc, ldsdst, rows, t);
#endif
}

// ============================================================================
// Kernel 0: convert the 4 weight matrices (fp32 [o][c]) to bf16 in workspace.
// grid = 256 blocks x 256 threads; 4 elements per thread.
// ============================================================================
__global__ __launch_bounds__(256) void wcvt_kernel(
    const float* __restrict__ wq, const float* __restrict__ wk,
    const float* __restrict__ wv, const float* __restrict__ wp,
    unsigned short* __restrict__ wb)
{
    const int t   = blockIdx.x * 256 + threadIdx.x;   // 0..65535
    const int mat = t >> 14;                          // 16384 threads per matrix
    const int idx = (t & 16383) * 4;
    const float* srcs[4] = { wq, wk, wv, wp };
    const float4 v = *(const float4*)&srcs[mat][idx];
    const unsigned long long pk =
        (unsigned long long)pack2bf(v.x, v.y) |
        ((unsigned long long)pack2bf(v.z, v.w) << 32);
    *(unsigned long long*)&wb[mat * (C_ * C_) + idx] = pk;
}

// ============================================================================
// Kernel 1: QKV projection.  grid = B*N*4, block = 256 (8 waves).
// ============================================================================
__global__ __launch_bounds__(256) void qkv_kernel(
    const float* __restrict__ x,
    const float* __restrict__ bq, const float* __restrict__ bk,
    const float* __restrict__ bv,
    const unsigned short* __restrict__ wb,     // bf16 weights [4][256][256]
    unsigned short* __restrict__ qws,
    unsigned short* __restrict__ kws,
    unsigned short* __restrict__ vws)
{
    __shared__ unsigned short xbuf[64 * LDSTR];   // 33280 B : X^T [pos][ch] bf16
    __shared__ unsigned short stag[8 * 256];      //  4096 B : per-wave D-tile staging

    const int blk   = blockIdx.x;
    const int pc    = blk & 3;
    const int n     = (blk >> 2) & (N_ - 1);
    const int b     = blk >> 8;
    const int pbase = pc * 64;
    const int t     = threadIdx.x;
    const int w     = t >> 5;
    const int lane  = t & 31;
    const int h     = lane >> 4;
    const int nn    = lane & 15;
    const long bn   = (long)b * N_ + n;

    // Load + transpose x[b, :, n, pbase..+64) -> xbuf[p][c]:
    // two channel rows per step -> packed dword LDS stores.
    {
        const int p4 = (t & 15) * 4;
        const int cb = (t >> 4) * 2;           // 0,2,..,30
#pragma unroll 2
        for (int i = 0; i < 8; ++i) {
            const int c = cb + i * 32;
            const float4 xa = *(const float4*)&x[(((long)b * C_ + c    ) * N_ + n) * K_ + pbase + p4];
            const float4 xb = *(const float4*)&x[(((long)b * C_ + c + 1) * N_ + n) * K_ + pbase + p4];
            *(unsigned int*)&xbuf[(p4 + 0) * LDSTR + c] = pack2bf(xa.x, xb.x);
            *(unsigned int*)&xbuf[(p4 + 1) * LDSTR + c] = pack2bf(xa.y, xb.y);
            *(unsigned int*)&xbuf[(p4 + 2) * LDSTR + c] = pack2bf(xa.z, xb.z);
            *(unsigned int*)&xbuf[(p4 + 3) * LDSTR + c] = pack2bf(xa.w, xb.w);
        }
    }
    __syncthreads();

    const float* Bs[3] = { bq, bk, bv };
    unsigned short* st = &stag[w * 256];

#pragma unroll 1
    for (int wgt = 0; wgt < 3; ++wgt) {
        const unsigned short* W = wb + wgt * (C_ * C_);   // bf16 [o][c]
        const float* bias = Bs[wgt];
#pragma unroll 1
        for (int oo = 0; oo < 2; ++oo) {
            const int ot    = w + oo * 8;
            const int obase = ot * 16;
            BF16x16 bfrag[8];
#pragma unroll
            for (int cs = 0; cs < 8; ++cs)
                bfrag[cs].v = load_b_frag(W + obase * C_ + cs * 32, C_, lane);
            const float bval = bias[obase + nn];

#pragma unroll 1
            for (int pt = 0; pt < 4; ++pt) {
                v8f acc = vzero();
#pragma unroll
                for (int cs = 0; cs < 8; ++cs) {
                    v16bf a = load_a_frag(&xbuf[(pt * 16) * LDSTR + cs * 32], LDSTR, lane);
                    acc = __builtin_amdgcn_wmma_f32_16x16x32_bf16(
                        false, a, false, bfrag[cs].v, (short)0, acc, false, false);
                }
                if (wgt < 2) {
                    // Q/K: [pos][ch]; repack via per-wave LDS staging -> b128 stores.
                    unsigned short* dst = (wgt == 0) ? qws : kws;
#pragma unroll
                    for (int r = 0; r < 8; ++r)
                        st[(r + 8 * h) * 16 + nn] = f2bf(acc[r] + bval);
                    const int row  = lane >> 1;
                    const int half = lane & 1;
                    U4 val = *(const U4*)&st[row * 16 + half * 8];
                    const int p = pbase + pt * 16 + row;
                    *(U4*)&dst[(bn * K_ + p) * C_ + obase + half * 8] = val;
                } else {
                    // V: [ch][pos]; 8 consecutive positions per lane -> b128.
                    const int o  = obase + nn;
                    const int p0 = pbase + pt * 16 + 8 * h;
                    U4 val = { pack2bf(acc[0] + bval, acc[1] + bval),
                               pack2bf(acc[2] + bval, acc[3] + bval),
                               pack2bf(acc[4] + bval, acc[5] + bval),
                               pack2bf(acc[6] + bval, acc[7] + bval) };
                    *(U4*)&vws[(bn * C_ + o) * K_ + p0] = val;
                }
            }
        }
    }
}

// ============================================================================
// Kernel 2: fused attention + output projection + residual.
// grid = B*N*8, block = 256 (8 waves); 32 query rows per block.
// LDS (49920 B): buf0 @0 (16640 B): Q -> P -> O [32][LDSTR] bf16
//                buf1 @16640 (33280 B): K -> S(f32) -> V
// ============================================================================
__global__ __launch_bounds__(256) void attn_kernel(
    const unsigned short* __restrict__ qws,
    const unsigned short* __restrict__ kws,
    const unsigned short* __restrict__ vws,
    const unsigned short* __restrict__ wpb,    // bf16 Wp [o][c]
    const float* __restrict__ bp,
    const float* __restrict__ x, float* __restrict__ out)
{
    __shared__ unsigned short smem[24960];
    unsigned short* buf0 = smem;
    unsigned short* buf1 = smem + 8320;
    float*          sbuf = (float*)(smem + 8320);

    const int blk   = blockIdx.x;
    const int kc    = blk & 7;
    const int n     = (blk >> 3) & (N_ - 1);
    const int b     = blk >> 9;
    const int kbase = kc * 32;
    const int t     = threadIdx.x;
    const int w     = t >> 5;
    const int lane  = t & 31;
    const int h     = lane >> 4;
    const int nn    = lane & 15;
    const long bn   = (long)b * N_ + n;

    // ---- Q chunk [32 pos][256 ch] -> buf0 via TDM ---------------------------
    tile_load(&qws[(bn * K_ + kbase) * C_], buf0, 32, t, w);

    // ---- Phase 1: S = Q^T K -------------------------------------------------
    const int kt = w >> 2;
    const int lt = w & 3;
    v8f sacc[4];
#pragma unroll 1
    for (int lc = 0; lc < 4; ++lc) {
        __syncthreads();
        tile_load(&kws[(bn * K_ + lc * 64) * C_], buf1, 64, t, w);
        __syncthreads();
        v8f acc = vzero();
#pragma unroll
        for (int cs = 0; cs < 8; ++cs) {
            v16bf a  = load_a_frag(&buf0[(kt * 16) * LDSTR + cs * 32], LDSTR, lane);
            v16bf bf = load_b_frag(&buf1[(lt * 16) * LDSTR + cs * 32], LDSTR, lane);
            acc = __builtin_amdgcn_wmma_f32_16x16x32_bf16(
                false, a, false, bf, (short)0, acc, false, false);
        }
        sacc[lc] = acc;
    }
    __syncthreads();

    // ---- Phase 2: softmax ---------------------------------------------------
    const float scale = 0.0625f;   // 1/sqrt(256)
#pragma unroll
    for (int lc = 0; lc < 4; ++lc)
#pragma unroll
        for (int r = 0; r < 8; ++r) {
            const int row = kt * 16 + r + 8 * h;
            const int col = lc * 64 + lt * 16 + nn;
            sbuf[row * LDSTR + col] = sacc[lc][r] * scale;
        }
    __syncthreads();
    {
        const int row = t >> 3;
        const int sub = t & 7;
        float vals[32];
        float mx = -1e30f;
#pragma unroll
        for (int j = 0; j < 32; ++j) {
            vals[j] = sbuf[row * LDSTR + sub * 32 + j];
            mx = fmaxf(mx, vals[j]);
        }
#pragma unroll
        for (int m2 = 1; m2 < 8; m2 <<= 1) mx = fmaxf(mx, __shfl_xor(mx, m2, 32));
        float sum = 0.0f;
#pragma unroll
        for (int j = 0; j < 32; ++j) { vals[j] = __expf(vals[j] - mx); sum += vals[j]; }
#pragma unroll
        for (int m2 = 1; m2 < 8; m2 <<= 1) sum += __shfl_xor(sum, m2, 32);
        const float inv = 1.0f / sum;
        __syncthreads();               // Q reads done; reuse buf0 for P
#pragma unroll
        for (int j = 0; j < 32; j += 2)
            *(unsigned int*)&buf0[row * LDSTR + sub * 32 + j] =
                pack2bf(vals[j] * inv, vals[j + 1] * inv);
    }
    __syncthreads();

    // ---- Phase 3: O = P V^T -------------------------------------------------
    const int ct  = w >> 1;
    const int kt2 = w & 1;
    v8f oacc[4];
#pragma unroll 1
    for (int cc = 0; cc < 4; ++cc) {
        __syncthreads();
        tile_load(&vws[(bn * C_ + cc * 64) * K_], buf1, 64, t, w);
        __syncthreads();
        v8f acc = vzero();
#pragma unroll
        for (int ls = 0; ls < 8; ++ls) {
            v16bf a  = load_a_frag(&buf0[(kt2 * 16) * LDSTR + ls * 32], LDSTR, lane);
            v16bf bf = load_b_frag(&buf1[(ct * 16) * LDSTR + ls * 32], LDSTR, lane);
            acc = __builtin_amdgcn_wmma_f32_16x16x32_bf16(
                false, a, false, bf, (short)0, acc, false, false);
        }
        oacc[cc] = acc;
    }
    __syncthreads();

    // ---- Phase 4: O -> buf0, y = Wp O + bp + x ------------------------------
#pragma unroll
    for (int cc = 0; cc < 4; ++cc)
#pragma unroll
        for (int r = 0; r < 8; ++r) {
            const int row = kt2 * 16 + r + 8 * h;
            const int col = cc * 64 + ct * 16 + nn;
            buf0[row * LDSTR + col] = f2bf(oacc[cc][r]);
        }
    __syncthreads();

#pragma unroll 1
    for (int oo = 0; oo < 2; ++oo) {
        const int ot    = w + oo * 8;
        const int obase = ot * 16;
        BF16x16 bfrag[8];
#pragma unroll
        for (int cs = 0; cs < 8; ++cs)
            bfrag[cs].v = load_b_frag(wpb + obase * C_ + cs * 32, C_, lane);
        const float bias = bp[obase + nn];

#pragma unroll 1
        for (int pt = 0; pt < 2; ++pt) {
            v8f acc = vzero();
#pragma unroll
            for (int cs = 0; cs < 8; ++cs) {
                v16bf a = load_a_frag(&buf0[(pt * 16) * LDSTR + cs * 32], LDSTR, lane);
                acc = __builtin_amdgcn_wmma_f32_16x16x32_bf16(
                    false, a, false, bfrag[cs].v, (short)0, acc, false, false);
            }
            // 8 consecutive output positions per lane -> 2x float4 load/store.
            const int o  = obase + nn;
            const int p0 = kbase + pt * 16 + 8 * h;
            const long gi = (((long)b * C_ + o) * N_ + n) * K_ + p0;
            const float4 x0 = *(const float4*)&x[gi];
            const float4 x1 = *(const float4*)&x[gi + 4];
            float4 r0, r1;
            r0.x = acc[0] + bias + x0.x;  r0.y = acc[1] + bias + x0.y;
            r0.z = acc[2] + bias + x0.z;  r0.w = acc[3] + bias + x0.w;
            r1.x = acc[4] + bias + x1.x;  r1.y = acc[5] + bias + x1.y;
            r1.z = acc[6] + bias + x1.z;  r1.w = acc[7] + bias + x1.w;
            *(float4*)&out[gi]     = r0;
            *(float4*)&out[gi + 4] = r1;
        }
    }
}

// ============================================================================
extern "C" void kernel_launch(void* const* d_in, const int* in_sizes, int n_in,
                              void* d_out, int out_size, void* d_ws, size_t ws_size,
                              hipStream_t stream) {
    const float* x  = (const float*)d_in[0];
    const float* wq = (const float*)d_in[1];
    const float* bq = (const float*)d_in[2];
    const float* wk = (const float*)d_in[3];
    const float* bk = (const float*)d_in[4];
    const float* wv = (const float*)d_in[5];
    const float* bv = (const float*)d_in[6];
    const float* wp = (const float*)d_in[7];
    const float* bp = (const float*)d_in[8];
    float* out = (float*)d_out;

    const size_t elems = (size_t)B_ * N_ * K_ * C_;   // 33,554,432 bf16 per tensor
    unsigned short* qws = (unsigned short*)d_ws;
    unsigned short* kws = qws + elems;
    unsigned short* vws = kws + elems;
    unsigned short* wb  = vws + elems;                // 4 x 65536 bf16 weights
                                                      // needs ws_size >= 192.5 MiB

    wcvt_kernel<<<dim3(256), dim3(256), 0, stream>>>(wq, wk, wv, wp, wb);
    qkv_kernel<<<dim3(B_ * N_ * 4), dim3(256), 0, stream>>>(
        x, bq, bk, bv, wb, qws, kws, vws);
    attn_kernel<<<dim3(B_ * N_ * 8), dim3(256), 0, stream>>>(
        qws, kws, vws, wb + 3 * (C_ * C_), bp, x, out);
}
